// CrossPoE_9096740733593
// MI455X (gfx1250) — matrix operand order repaired and checked by reading, not compiled, not run on hardware
//
#include <hip/hip_runtime.h>

// ---------------------------------------------------------------------------
// CrossPoE VAE forward for MI455X (gfx1250, wave32, WMMA bf16).
// GEMM tiles are staged into LDS in *WMMA fragment order* so every lane reads
// its whole 16-elem bf16 fragment as one contiguous 32B LDS access
// (2x ds_load_b128). 128x64 tile per 128-thread block; each wave computes
// 32x64 = 8 WMMA accumulators per K-step with software-pipelined global
// loads. fp32->bf16 uses native __bf16 casts so clang emits hardware
// v_cvt (packed) instead of manual RNE bit-twiddling.
// ---------------------------------------------------------------------------

#define B_ROWS   2048
#define LATENT   48
#define HID      1024
#define D_RNA    4652
#define D_MIRNA  524
#define D_METHYL 37482

#define TM 128
#define TN 64
#define TK 32
#define NTHREADS 128

typedef __attribute__((ext_vector_type(16))) __bf16 v16bf;
typedef __attribute__((ext_vector_type(2)))  __bf16 bf16x2;
typedef __attribute__((ext_vector_type(8)))  float  v8f;

__device__ __forceinline__ unsigned pack2(float x, float y) {
    bf16x2 v;
    v[0] = (__bf16)x;          // native convert -> v_cvt_pk_bf16_f32 class
    v[1] = (__bf16)y;
    return __builtin_bit_cast(unsigned, v);
}

// ---------------------------------------------------------------------------
// C[M,N] = act(A[M,K] @ W[K,N] + bias[N]) (* mask column)
// Fragment-ordered LDS:
//   sAf[strip][lane][e] : A fragment for rows strip*16..+15.  Lane L holds
//     row (L&15), K = (L>>4)*8 + {0..7} (e=0..7) and +16 (e=8..15); each run
//     of 8 elements is 8 *consecutive* K values -> 16B-contiguous staging.
//   sBf[nt][lane][e]    : B fragment for cols nt*16..+15.  Lane L holds
//     col (L&15), K = (L>>4)*8 + {0..7} (e=0..7) and +16 (e=8..15), same as sAf.
// ---------------------------------------------------------------------------
__global__ __launch_bounds__(NTHREADS)
void gemm_bf16_wmma(const float* __restrict__ A,
                    const float* __restrict__ W,
                    const float* __restrict__ bias,
                    float* __restrict__ C,
                    int M, int N, int K,
                    int relu,
                    const int* __restrict__ mask, int maskIdx)
{
    __shared__ __align__(32) unsigned short sAf[TM / 16][32][16]; // 8 KB
    __shared__ __align__(32) unsigned short sBf[TN / 16][32][16]; // 4 KB

    const int tid  = threadIdx.x;
    const int lane = tid & 31;
    const int wave = tid >> 5;                  // 0..3
    const int m0   = blockIdx.y * TM;           // M assumed multiple of TM (2048)
    const int n0   = blockIdx.x * TN;

    // staging registers (software pipeline)
    float2 aReg[4][4];                          // 4 tasks x 8 consecutive floats
    float  bReg[4][4];                          // 4 tasks x 4-row column group

    // A task j: t = j*128+tid; r = t>>2 (row 0..127), g = t&3 (8-wide k group)
    auto loadA = [&](int k0) {
        #pragma unroll
        for (int j = 0; j < 4; ++j) {
            int t = j * NTHREADS + tid;
            int r = t >> 2, g = t & 3;
            const float* row = A + (size_t)(m0 + r) * K + (k0 + g * 8);
            if (k0 + TK <= K) {                 // fast path: float2 (K is even)
                const float2* p = reinterpret_cast<const float2*>(row);
                aReg[j][0] = p[0]; aReg[j][1] = p[1];
                aReg[j][2] = p[2]; aReg[j][3] = p[3];
            } else {                            // K edge: guarded scalar
                #pragma unroll
                for (int q = 0; q < 4; ++q) {
                    int kk = k0 + g * 8 + q * 2;
                    float2 v;
                    v.x = (kk     < K) ? row[q * 2]     : 0.0f;
                    v.y = (kk + 1 < K) ? row[q * 2 + 1] : 0.0f;
                    aReg[j][q] = v;
                }
            }
        }
    };

    // B task j: t = j*128+tid; c = t&63 (col), kq = t>>6 (4-row k group)
    auto loadB = [&](int k0) {
        #pragma unroll
        for (int j = 0; j < 4; ++j) {
            int t = j * NTHREADS + tid;
            int c = t & 63, kq = t >> 6;
            int gc = n0 + c;
            #pragma unroll
            for (int i = 0; i < 4; ++i) {
                int gk = k0 + kq * 4 + i;
                bReg[j][i] = (gk < K && gc < N) ? W[(size_t)gk * N + gc] : 0.0f;
            }
        }
    };

    auto stage = [&]() {
        #pragma unroll
        for (int j = 0; j < 4; ++j) {           // A: one ds_store_b128 per task
            int t = j * NTHREADS + tid;
            int r = t >> 2, g = t & 3;
            int strip = r >> 4;
            int ln    = ((g & 1) << 4) + (r & 15);
            int eb    = (g >> 1) << 3;
            uint4 w;
            w.x = pack2(aReg[j][0].x, aReg[j][0].y);
            w.y = pack2(aReg[j][1].x, aReg[j][1].y);
            w.z = pack2(aReg[j][2].x, aReg[j][2].y);
            w.w = pack2(aReg[j][3].x, aReg[j][3].y);
            *reinterpret_cast<uint4*>(&sAf[strip][ln][eb]) = w;
        }
        #pragma unroll
        for (int j = 0; j < 4; ++j) {           // B: one ds_store_b64 per task
            int t = j * NTHREADS + tid;
            int c = t & 63, kq = t >> 6;
            int kr = kq * 4;
            int nt = c >> 4;
            int ln = (((kr >> 3) & 1) << 4) + (c & 15);
            int eb = (kr & 7) + ((kr >> 4) << 3);
            uint2 w;
            w.x = pack2(bReg[j][0], bReg[j][1]);
            w.y = pack2(bReg[j][2], bReg[j][3]);
            *reinterpret_cast<uint2*>(&sBf[nt][ln][eb]) = w;
        }
    };

    v8f acc[2][4] = {};                         // 2 row-strips x 4 col-tiles

    loadA(0);
    loadB(0);

    #pragma unroll 1
    for (int k0 = 0; k0 < K; k0 += TK) {
        stage();
        __syncthreads();

        int kn = k0 + TK;
        if (kn < K) {                           // pipeline next tile
            loadA(kn);
            loadB(kn);
            if (kn + TK < K) {                  // L2 warm for tile after next
                __builtin_prefetch(A + (size_t)(m0 + (tid & (TM - 1))) * K + kn + TK, 0, 0);
                __builtin_prefetch(W + (size_t)(kn + TK + (tid >> 2)) * N + n0, 0, 0);
            }
        }

        // fragment reads: contiguous 32B per lane -> 2x ds_load_b128 each
        v16bf a0 = *reinterpret_cast<const v16bf*>(sAf[wave * 2 + 0][lane]);
        v16bf a1 = *reinterpret_cast<const v16bf*>(sAf[wave * 2 + 1][lane]);
        #pragma unroll
        for (int nt = 0; nt < 4; ++nt) {
            v16bf b = *reinterpret_cast<const v16bf*>(sBf[nt][lane]);
            acc[0][nt] = __builtin_amdgcn_wmma_f32_16x16x32_bf16(
                false, a0, false, b, (short)0, acc[0][nt], false, false);
            acc[1][nt] = __builtin_amdgcn_wmma_f32_16x16x32_bf16(
                false, a1, false, b, (short)0, acc[1][nt], false, false);
        }
        __syncthreads();
    }

    // ---- epilogue: bias (+ReLU) (+mask column); C/D f32 VGPR layout -------
    #pragma unroll
    for (int si = 0; si < 2; ++si) {
        int rbase = m0 + (wave * 2 + si) * 16 + (lane >> 4) * 8;
        #pragma unroll
        for (int nt = 0; nt < 4; ++nt) {
            int gc = n0 + nt * 16 + (lane & 15);
            if (gc >= N) continue;
            float bv = bias ? bias[gc] : 0.0f;
            #pragma unroll
            for (int e = 0; e < 8; ++e) {
                int gr = rbase + e;
                float v = acc[si][nt][e] + bv;
                if (relu) v = fmaxf(v, 0.0f);
                if (mask) v *= (mask[gr * 3 + maskIdx] != 0) ? 1.0f : 0.0f;
                C[(size_t)gr * N + gc] = v;
            }
        }
    }
}

// ---------------------------------------------------------------------------
// PoE combine + reparameterization (deterministic hash-based eps).
// ---------------------------------------------------------------------------
__device__ __forceinline__ unsigned pcg_hash(unsigned x) {
    x = x * 747796405u + 2891336453u;
    unsigned w = ((x >> ((x >> 28) + 4u)) ^ x) * 277803737u;
    return (w >> 22) ^ w;
}

__device__ __forceinline__ float rng_normal(unsigned i) {
    unsigned a = pcg_hash(i * 2u + 1u);
    unsigned b = pcg_hash(i * 2u + 2u);
    float u1 = ((float)a + 1.0f) * 2.3283064e-10f;   // (0,1]
    float u2 = (float)b * 2.3283064e-10f;            // [0,1)
    return sqrtf(fmaxf(-2.0f * __logf(u1), 0.0f)) * __cosf(6.2831853f * u2);
}

__global__ __launch_bounds__(256)
void poe_combine(const float* __restrict__ mu0, const float* __restrict__ lv0,
                 const float* __restrict__ mu1, const float* __restrict__ lv1,
                 const float* __restrict__ mu2, const float* __restrict__ lv2,
                 const int* __restrict__ mask,
                 float* __restrict__ out_mu, float* __restrict__ out_lv,
                 float* __restrict__ out_z,  float* __restrict__ ws_z,
                 int n)
{
    int i = blockIdx.x * blockDim.x + threadIdx.x;
    if (i >= n) return;
    int b = i / LATENT;

    const float* mus[3] = { mu0, mu1, mu2 };
    const float* lvs[3] = { lv0, lv1, lv2 };

    float ps = 1.0f, wm = 0.0f;
    #pragma unroll
    for (int j = 0; j < 3; ++j) {
        float m    = (mask[b * 3 + j] != 0) ? 1.0f : 0.0f;
        float lv   = lvs[j][i] * m;
        float mu   = mus[j][i] * m;
        float prec = __expf(-lv) * m;
        ps += prec;
        wm += prec * mu;
    }
    float var    = 1.0f / (ps + 1e-8f);
    float logvar = __logf(var + 1e-8f);
    float mu_poe = var * wm;
    float eps    = rng_normal((unsigned)i);
    float z      = mu_poe + eps * __expf(0.5f * logvar);

    out_mu[i] = mu_poe;
    out_lv[i] = logvar;
    out_z[i]  = z;
    ws_z[i]   = z;
}

// ---------------------------------------------------------------------------
// Host-side orchestration
// ---------------------------------------------------------------------------
static inline void launch_gemm(const float* A, const float* W, const float* bias,
                               float* C, int M, int N, int K, int relu,
                               const int* mask, int maskIdx, hipStream_t s)
{
    dim3 grid((N + TN - 1) / TN, (M + TM - 1) / TM);
    gemm_bf16_wmma<<<grid, NTHREADS, 0, s>>>(A, W, bias, C, M, N, K, relu, mask, maskIdx);
}

extern "C" void kernel_launch(void* const* d_in, const int* in_sizes, int n_in,
                              void* d_out, int out_size, void* d_ws, size_t ws_size,
                              hipStream_t stream)
{
    (void)in_sizes; (void)n_in; (void)out_size; (void)ws_size;

    static const int DIMS[3] = { D_RNA, D_MIRNA, D_METHYL };

    const float* X[3] = { (const float*)d_in[0], (const float*)d_in[1],
                          (const float*)d_in[2] };
    const int* mask   = (const int*)d_in[3];

    // workspace layout (floats)
    const size_t F = (size_t)B_ROWS * LATENT;          // 98304
    float* ws     = (float*)d_ws;
    float* muW[3] = { ws + 0 * F, ws + 1 * F, ws + 2 * F };
    float* lvW[3] = { ws + 3 * F, ws + 4 * F, ws + 5 * F };
    float* zW     = ws + 6 * F;
    float* H      = ws + 7 * F;                        // 2048x1024, reused

    // output: mu_poe | logvar_poe | z | recon_rna | recon_mirna | recon_methyl
    float* out    = (float*)d_out;
    float* out_mu = out;
    float* out_lv = out + F;
    float* out_z  = out + 2 * F;
    float* out_rec[3];
    out_rec[0] = out + 3 * F;
    out_rec[1] = out_rec[0] + (size_t)B_ROWS * D_RNA;
    out_rec[2] = out_rec[1] + (size_t)B_ROWS * D_MIRNA;

    // ---------------- encoders ----------------
    for (int i = 0; i < 3; ++i) {
        const int base = 4 + i * 10;
        const float* enc_w1  = (const float*)d_in[base + 0];
        const float* enc_b1  = (const float*)d_in[base + 1];
        const float* enc_wmu = (const float*)d_in[base + 2];
        const float* enc_bmu = (const float*)d_in[base + 3];
        const float* enc_wlv = (const float*)d_in[base + 4];
        const float* enc_blv = (const float*)d_in[base + 5];

        launch_gemm(X[i], enc_w1, enc_b1, H, B_ROWS, HID, DIMS[i], 1, nullptr, 0, stream);
        launch_gemm(H, enc_wmu, enc_bmu, muW[i], B_ROWS, LATENT, HID, 0, nullptr, 0, stream);
        launch_gemm(H, enc_wlv, enc_blv, lvW[i], B_ROWS, LATENT, HID, 0, nullptr, 0, stream);
    }

    // ---------------- PoE + reparameterization ----------------
    {
        int n = (int)F;
        poe_combine<<<(n + 255) / 256, 256, 0, stream>>>(
            muW[0], lvW[0], muW[1], lvW[1], muW[2], lvW[2], mask,
            out_mu, out_lv, out_z, zW, n);
    }

    // ---------------- decoders ----------------
    for (int i = 0; i < 3; ++i) {
        const int base = 4 + i * 10;
        const float* dec_w1 = (const float*)d_in[base + 6];
        const float* dec_b1 = (const float*)d_in[base + 7];
        const float* dec_w2 = (const float*)d_in[base + 8];
        const float* dec_b2 = (const float*)d_in[base + 9];

        launch_gemm(zW, dec_w1, dec_b1, H, B_ROWS, HID, LATENT, 1, nullptr, 0, stream);
        launch_gemm(H, dec_w2, dec_b2, out_rec[i], B_ROWS, DIMS[i], HID, 0, mask, i, stream);
    }
}
